// QNet_51135880626943
// MI455X (gfx1250) — matrix operand order repaired and checked by reading, not compiled, not run on hardware
//
#include <hip/hip_runtime.h>

typedef __attribute__((ext_vector_type(16))) _Float16 v16h;
typedef __attribute__((ext_vector_type(8)))  float    v8f;

#define NODES_PER_GRAPH 256
#define DLAT 64
#define KDIM 128   // 2*D
#define HDIM 128
#define NEGF (-3.402823466e+38f)

// ---------------------------------------------------------------------------
// Kernel 1: W1 [K=128][H=128] f32 row-major  ->  W1^T [H][K] f16 in workspace.
// Lets the main kernel stage B-fragments (column-contiguous K) with coalesced
// reads and conflict-free LDS writes.
// ---------------------------------------------------------------------------
__global__ __launch_bounds__(256) void qnet_prep_w1t(const float* __restrict__ W1,
                                                     _Float16* __restrict__ w1t) {
  const int idx = blockIdx.x * 256 + threadIdx.x;  // 0..16383
  const int n = idx >> 7;    // H index (row of W1^T)
  const int k = idx & 127;   // K index (col of W1^T)
  w1t[idx] = (_Float16)W1[k * HDIM + n];
}

// ---------------------------------------------------------------------------
// Kernel 2: one workgroup per graph (256 nodes). Fused:
//   concat(embed, graph_embed) @ W1 + b1 -> relu -> @W2 + b2 -> raw_pred
//   banned mask -> per-graph argmax (first max wins) -> actions/values
// 512 threads = 16 waves; wave w computes rows [w*16, w*16+16) x all 128 cols
// with v_wmma_f32_16x16x32_f16 (8 N-tiles, 4 K-steps = 32 WMMAs / wave).
// ---------------------------------------------------------------------------
__global__ __launch_bounds__(512) void qnet_fused(
    const float* __restrict__ embed,        // [N, 64]
    const float* __restrict__ graph_embed,  // [B, 64]
    const int*   __restrict__ prefix_sum,   // [B]
    const int*   __restrict__ banned,       // [N]
    const _Float16* __restrict__ w1t,       // [H=128][K=128] f16 (from ws)
    const float* __restrict__ b1,           // [128]
    const float* __restrict__ w2,           // [128]
    const float* __restrict__ b2,           // [1]
    float* __restrict__ out_actions,        // [B]
    float* __restrict__ out_values,         // [B]
    float* __restrict__ out_raw)            // [N]
{
  __shared__ __align__(32) _Float16 sW1T[HDIM * KDIM]; // 32 KB
  __shared__ __align__(16) float    sGE[DLAT];
  __shared__ float sB1[HDIM];
  __shared__ float sW2[HDIM];
  __shared__ float sRaw[NODES_PER_GRAPH];
  __shared__ float sQ[NODES_PER_GRAPH];
  __shared__ int   sIdx[NODES_PER_GRAPH];

  const int g = blockIdx.x;
  const int t = threadIdx.x;
  const int start = (g == 0) ? 0 : prefix_sum[g - 1];

  // --- stage W1^T (f16), biases, W2, graph embed into LDS -------------------
  {
    const uint4* __restrict__ src = (const uint4*)w1t;  // 2048 x 16B
    uint4* dst = (uint4*)sW1T;
#pragma unroll
    for (int i = 0; i < 4; ++i) dst[t + i * 512] = src[t + i * 512];
  }
  if (t < HDIM) { sB1[t] = b1[t]; sW2[t] = w2[t]; }
  if (t < DLAT) { sGE[t] = graph_embed[g * DLAT + t]; }
  __syncthreads();

  const int wave = t >> 5;
  const int lane = t & 31;
  const int lm   = lane & 15;   // row (A) / col (B) within 16-tile
  const int lhi  = lane >> 4;   // K-half selector per WMMA layout
  const int node = start + wave * 16 + lm;

  v8f acc[8] = {};  // 8 N-tiles of 16x16 f32

  const float* __restrict__ arow = embed + (size_t)node * DLAT;

#pragma unroll
  for (int ks = 0; ks < 4; ++ks) {
    const int kb = ks * 32;
    const int k0 = kb + lhi * 8;  // A: elems 0-7 -> K k0..k0+7 ; 8-15 -> k0+16..k0+23
    v16h a;
    if (ks < 2) {  // K < 64: node embedding, streamed once from HBM
      const float4 x0 = *(const float4*)(arow + k0);
      const float4 x1 = *(const float4*)(arow + k0 + 4);
      const float4 y0 = *(const float4*)(arow + k0 + 16);
      const float4 y1 = *(const float4*)(arow + k0 + 20);
      a[0]  = (_Float16)x0.x; a[1]  = (_Float16)x0.y; a[2]  = (_Float16)x0.z; a[3]  = (_Float16)x0.w;
      a[4]  = (_Float16)x1.x; a[5]  = (_Float16)x1.y; a[6]  = (_Float16)x1.z; a[7]  = (_Float16)x1.w;
      a[8]  = (_Float16)y0.x; a[9]  = (_Float16)y0.y; a[10] = (_Float16)y0.z; a[11] = (_Float16)y0.w;
      a[12] = (_Float16)y1.x; a[13] = (_Float16)y1.y; a[14] = (_Float16)y1.z; a[15] = (_Float16)y1.w;
    } else {       // K >= 64: broadcast graph embedding from LDS
      const int kg = k0 - DLAT;
#pragma unroll
      for (int e = 0; e < 8; ++e) {
        a[e]     = (_Float16)sGE[kg + e];
        a[8 + e] = (_Float16)sGE[kg + 16 + e];
      }
    }

    const int kbb = kb + lhi * 8;   // B: elems 0-7 -> K kbb..kbb+7 ; 8-15 -> kbb+16..kbb+23
#pragma unroll
    for (int j = 0; j < 8; ++j) {
      const int n = j * 16 + lm;
      v16h bfrag;
      {
        const uint4 lo = *(const uint4*)&sW1T[n * KDIM + kbb];
        const uint4 hi = *(const uint4*)&sW1T[n * KDIM + kbb + 16];
        uint4* bp = (uint4*)&bfrag;
        bp[0] = lo; bp[1] = hi;
      }
      acc[j] = __builtin_amdgcn_wmma_f32_16x16x32_f16(
          /*neg_a=*/false, a, /*neg_b=*/false, bfrag,
          /*c_mod=*/(short)0, acc[j], /*reuse_a=*/false, /*reuse_b=*/false);
    }
  }

  // --- epilogue: relu(h + b1) . W2, reduce across the 16 lanes of each row --
  const float b2v = b2[0];
#pragma unroll
  for (int r = 0; r < 8; ++r) {
    float p = 0.0f;
#pragma unroll
    for (int j = 0; j < 8; ++j) {
      const int col = j * 16 + lm;          // this lane's N column in tile j
      float h = acc[j][r] + sB1[col];
      h = h > 0.0f ? h : 0.0f;
      p += h * sW2[col];
    }
#pragma unroll
    for (int m = 1; m < 16; m <<= 1) p += __shfl_xor(p, m, 32);  // stays in 16-lane group
    if (lm == 0) sRaw[wave * 16 + lhi * 8 + r] = p + b2v;  // C layout: M=r (lo), M=8+r (hi)
  }
  __syncthreads();

  // --- coalesced raw_pred store + banned mask ------------------------------
  if (t < NODES_PER_GRAPH) {
    const float raw = sRaw[t];
    out_raw[start + t] = raw;
    sQ[t]   = banned[start + t] ? NEGF : raw;
    sIdx[t] = t;
  }
  __syncthreads();

  // --- per-graph argmax, first max wins (min index among ties) -------------
#pragma unroll
  for (int s = 128; s > 0; s >>= 1) {
    if (t < s) {
      const float v2 = sQ[t + s];
      const int   i2 = sIdx[t + s];
      if (v2 > sQ[t] || (v2 == sQ[t] && i2 < sIdx[t])) { sQ[t] = v2; sIdx[t] = i2; }
    }
    __syncthreads();
  }
  if (t == 0) {
    out_actions[g] = (float)sIdx[0];  // already local (gidx - start)
    out_values[g]  = sQ[0];
  }
}

// ---------------------------------------------------------------------------
extern "C" void kernel_launch(void* const* d_in, const int* in_sizes, int n_in,
                              void* d_out, int out_size, void* d_ws, size_t ws_size,
                              hipStream_t stream) {
  const float* embed       = (const float*)d_in[0];
  const float* graph_embed = (const float*)d_in[1];
  const int*   prefix_sum  = (const int*)d_in[2];
  const int*   banned      = (const int*)d_in[3];
  const float* W1          = (const float*)d_in[4];
  const float* b1          = (const float*)d_in[5];
  const float* W2          = (const float*)d_in[6];
  const float* b2          = (const float*)d_in[7];

  const int B = in_sizes[2];  // number of graphs (4096)

  _Float16* w1t = (_Float16*)d_ws;  // 32 KB scratch

  float* out_actions = (float*)d_out;
  float* out_values  = out_actions + B;
  float* out_raw     = out_values + B;

  qnet_prep_w1t<<<(HDIM * KDIM) / 256, 256, 0, stream>>>(W1, w1t);
  qnet_fused<<<B, 512, 0, stream>>>(embed, graph_embed, prefix_sum, banned,
                                    w1t, b1, W2, b2,
                                    out_actions, out_values, out_raw);
}
